// TransformerSeq2Seq_84482006712426
// MI455X (gfx1250) — matrix operand order repaired and checked by reading, compiled, not run on hardware
//
#include <hip/hip_runtime.h>
#include <math.h>

// ---------------- model constants (fixed by reference) ----------------
#define MB      4
#define SLEN    1024          // SS == ST == 1024
#define DMODEL  512
#define NHEAD   8
#define DHEAD   64
#define DFF     2048
#define NVOCAB  32000
#define NLAYER  6
#define NTOK    (MB * SLEN)   // 4096 tokens per stream

// ---------------- WMMA GEMM tiling ----------------
#define BM 128                // block tile M
#define BN 64                 // block tile N
#define BK 32                 // K step (one WMMA K)
#define LDA_S (BK + 8)        // LDS row stride (halves), padded (TDM pad matches)
#define LDB_S (BN + 8)

typedef __attribute__((ext_vector_type(16))) _Float16 v16h;
typedef __attribute__((ext_vector_type(8)))  _Float16 h8;
typedef __attribute__((ext_vector_type(8)))  float    v8f;
typedef __attribute__((ext_vector_type(4)))  unsigned int u32x4;
typedef __attribute__((ext_vector_type(8)))  int      i32x8;
typedef __attribute__((ext_vector_type(4)))  int      i32x4;

// Tensor Data Mover path (6-arg builtin: g0, g1, g2, g3, extra group, cpol),
// guarded so the build cannot break if the builtin is absent.
#if defined(__has_builtin)
#if __has_builtin(__builtin_amdgcn_tensor_load_to_lds) && \
    __has_builtin(__builtin_amdgcn_s_wait_tensorcnt)
#define HAVE_TDM 1
#endif
#endif
#ifndef HAVE_TDM
#define HAVE_TDM 0
#endif

// ----------------------------------------------------------------------
// Fragment loaders, per CDNA5 ISA 7.12.2 VGPR layouts (wave32).
// A 16x32 f16: lane&15 = M row; VGPR v holds
//   K = (v>=4 ? 16:0) + (lane>>4)*8 + (v&3)*2 + {0,1}
// ----------------------------------------------------------------------
__device__ __forceinline__ v16h frag_a_lds(const _Float16* t, int lane) {
  const _Float16* p = t + (lane & 15) * LDA_S;
  int kg = (lane >> 4) << 3;
  v16h a;
#pragma unroll
  for (int vv = 0; vv < 8; ++vv) {
    int kb = ((vv & 4) << 2) + kg + ((vv & 3) << 1);
    a[2 * vv]     = p[kb];
    a[2 * vv + 1] = p[kb + 1];
  }
  return a;
}

// B 32x16 f16: lane&15 = N col; VGPR v holds
//   K = (v>=4 ? 16:0) + (lane>>4)*8 + (v&3)*2 + {0,1}
__device__ __forceinline__ v16h frag_b_lds(const _Float16* t, int lane) {
  const _Float16* p = t + (lane & 15);
  int kg = (lane >> 4) << 3;
  v16h b;
#pragma unroll
  for (int vv = 0; vv < 8; ++vv) {
    int kb = ((vv & 4) << 2) + kg + ((vv & 3) << 1);
    b[2 * vv]     = p[kb * LDB_S];
    b[2 * vv + 1] = p[(kb + 1) * LDB_S];
  }
  return b;
}

__device__ __forceinline__ float gelu_exact(float x) {
  return 0.5f * x * (1.0f + erff(x * 0.70710678118654752f));
}

// ----------------------------------------------------------------------
// Generic WMMA GEMM, all-f16 operands, f32 accumulate:
//   out = act(alpha * A@B(^T) + bias) + residual
// bmode 0: plain.  bmode 1: attn scores per z=(b*H+h) (transB, C=[z,M,N]).
// bmode 2: attn ctx per z (A=probs[z], B=v[b,:,h], C=ctx[b,:,h]).
// 8 waves; each wave owns a 32x32 output tile -> 4 WMMAs / K-step.
// A tile (BM x BK) staged by TDM when available; B tile by vector copy.
// All M,N,K divide the tile sizes for every GEMM in this model.
// ----------------------------------------------------------------------
__global__ __launch_bounds__(256)
void gemm_wmma_kernel(const _Float16* __restrict__ A, const _Float16* __restrict__ Bm,
                      const float* __restrict__ bias, const float* __restrict__ Res,
                      float* __restrict__ Cf, _Float16* __restrict__ Ch,
                      int M, int N, int K, int lda, int ldb, int ldc,
                      int bmode, int transB, int act, float alpha)
{
  __shared__ _Float16 As[BM * LDA_S];
  __shared__ _Float16 Bs[BK * LDB_S];

  size_t ofsA = 0, ofsB = 0, ofsC = 0;
  if (bmode != 0) {
    int z = blockIdx.z;
    int b = z >> 3, h = z & 7;
    if (bmode == 1) {
      ofsA = (size_t)b * M * lda + (size_t)h * DHEAD;
      ofsB = (size_t)b * N * ldb + (size_t)h * DHEAD;
      ofsC = (size_t)z * M * N;
    } else {
      ofsA = (size_t)z * M * K;
      ofsB = (size_t)b * K * ldb + (size_t)h * DHEAD;
      ofsC = (size_t)b * M * ldc + (size_t)h * DHEAD;
    }
  }

  const int m0   = blockIdx.y * BM;
  const int n0   = blockIdx.x * BN;
  const int tid  = threadIdx.x;
  const int lane = tid & 31;
  const int wave = tid >> 5;
  const int wm   = (wave & 3) * 32;   // wave M offset (4 waves over 128)
  const int wn   = (wave >> 2) * 32;  // wave N offset (2 waves over 64)

  // cooperative copy coordinates (contiguous halves per thread)
  const int ar  = tid >> 1,  ac  = (tid & 1) << 4;   // A fallback: 128x32, 16 halves/thr
  const int bk0 = tid >> 3,  bn0 = (tid & 7) << 3;   // B normal: 32x64, 8 halves/thr
  const int tn  = tid >> 2,  tk  = (tid & 3) << 3;   // B transposed source

  v8f acc00 = {}, acc01 = {}, acc10 = {}, acc11 = {};

  for (int k0 = 0; k0 < K; k0 += BK) {
    // ---------------- stage A tile (BM x BK halves) ----------------
#if HAVE_TDM
    if (wave == 0) {
      // Tensor DMA descriptor (ISA §8.3/§8.4): 2D tile, data_size=2B,
      // LDS padding 4 DWORDs every 16 DWORDs -> row stride BK+8 halves.
      unsigned long long ga =
          (unsigned long long)(const void*)(A + ofsA + (size_t)m0 * lda + k0);
      u32x4 g0;
      g0[0] = 1u;                                   // count=1, user descriptor
      g0[1] = (unsigned)(size_t)As;                 // lds_addr (bytes)
      g0[2] = (unsigned)(ga & 0xFFFFFFFFu);         // global_addr[31:0]
      g0[3] = (unsigned)((ga >> 32) & 0x01FFFFFFu)  // global_addr[56:32]
              | (2u << 30);                         // type=2 ("image")
      i32x8 g1;
      g1[0] = (int)(0x10000u          // data_size=1 -> 2 bytes
              | (1u << 20)            // pad_enable
              | (3u << 22)            // pad_interval code 3 -> every 16 DWORDs (64B)
              | (3u << 25));          // pad_amount  code 3 -> 4 DWORDs (8 halves)
      g1[1] = (int)((unsigned)BK << 16);            // tensor_dim0 = BK
      g1[2] = (int)((unsigned)BM << 16);            // tensor_dim0>>16=0 | tensor_dim1=BM
      g1[3] = (int)((unsigned)BK << 16);            // tensor_dim1>>16=0 | tile_dim0=BK
      g1[4] = BM;                                   // tile_dim1=BM, tile_dim2=0
      g1[5] = lda;                                  // tensor_dim0_stride (elements)
      g1[6] = 0;
      g1[7] = 0;
      i32x4 z4 = {0, 0, 0, 0};
      i32x8 z8 = {0, 0, 0, 0, 0, 0, 0, 0};
      __builtin_amdgcn_tensor_load_to_lds(g0, g1, z4, z4, z8, 0);
    }
#else
    {
      const _Float16* src = A + ofsA + (size_t)(m0 + ar) * lda + (k0 + ac);
      *reinterpret_cast<h8*>(&As[ar * LDA_S + ac])     = *reinterpret_cast<const h8*>(src);
      *reinterpret_cast<h8*>(&As[ar * LDA_S + ac + 8]) = *reinterpret_cast<const h8*>(src + 8);
      if (k0 + BK < K) __builtin_prefetch(src + BK, 0, 1);
    }
#endif
    // ---------------- stage B tile (BK x BN halves) ----------------
    if (!transB) {
      const _Float16* src = Bm + ofsB + (size_t)(k0 + bk0) * ldb + (n0 + bn0);
      *reinterpret_cast<h8*>(&Bs[bk0 * LDB_S + bn0]) = *reinterpret_cast<const h8*>(src);
      if (k0 + BK < K) __builtin_prefetch(src + BK * (size_t)ldb, 0, 1);
    } else {
      const _Float16* src = Bm + ofsB + (size_t)(n0 + tn) * ldb + (k0 + tk);
      h8 f = *reinterpret_cast<const h8*>(src);
#pragma unroll
      for (int j = 0; j < 8; ++j) Bs[(tk + j) * LDB_S + tn] = f[j];
      if (k0 + BK < K) __builtin_prefetch(src + BK, 0, 1);
    }
#if HAVE_TDM
    __builtin_amdgcn_s_wait_tensorcnt(0);   // no-op for waves with TENSORcnt==0
#endif
    __syncthreads();

    v16h a0 = frag_a_lds(As + (size_t)wm * LDA_S, lane);
    v16h a1 = frag_a_lds(As + (size_t)(wm + 16) * LDA_S, lane);
    v16h b0 = frag_b_lds(Bs + wn, lane);
    v16h b1 = frag_b_lds(Bs + wn + 16, lane);
    acc00 = __builtin_amdgcn_wmma_f32_16x16x32_f16(false, a0, false, b0, (short)0, acc00, false, false);
    acc01 = __builtin_amdgcn_wmma_f32_16x16x32_f16(false, a0, false, b1, (short)0, acc01, false, false);
    acc10 = __builtin_amdgcn_wmma_f32_16x16x32_f16(false, a1, false, b0, (short)0, acc10, false, false);
    acc11 = __builtin_amdgcn_wmma_f32_16x16x32_f16(false, a1, false, b1, (short)0, acc11, false, false);
    __syncthreads();
  }

  // C/D layout: VGPR v -> row = 16*sub_m + (lane>>4)*8 + v, col = 16*sub_n + (lane&15)
  const int ncol = lane & 15;
  const int mrow = (lane >> 4) << 3;
  const v8f* accs[4] = { &acc00, &acc01, &acc10, &acc11 };
#pragma unroll
  for (int s = 0; s < 4; ++s) {
    int rbase = m0 + wm + ((s >> 1) << 4) + mrow;
    int cbase = n0 + wn + ((s & 1) << 4) + ncol;
    const v8f& a = *accs[s];
#pragma unroll
    for (int vv = 0; vv < 8; ++vv) {
      int r = rbase + vv;
      float o = a[vv] * alpha;
      if (bias) o += bias[cbase];
      if (act == 1) o = gelu_exact(o);
      size_t idx = ofsC + (size_t)r * ldc + cbase;
      if (Ch) {
        Ch[idx] = (_Float16)o;
      } else {
        if (Res) o += Res[idx];
        Cf[idx] = o;
      }
    }
  }
}

// ----------------------------------------------------------------------
// f32 -> f16 cast (weights, once per launch)
// ----------------------------------------------------------------------
__global__ __launch_bounds__(256)
void cast_f16_kernel(const float* __restrict__ in, _Float16* __restrict__ out, long long n)
{
  long long i = (long long)blockIdx.x * 256 + threadIdx.x;
  if (i < n) out[i] = (_Float16)in[i];
}

// ----------------------------------------------------------------------
// Embedding gather + positional add (f32 residual stream)
// ----------------------------------------------------------------------
__global__ __launch_bounds__(256)
void embed_kernel(const int* __restrict__ ids, const float* __restrict__ emb,
                  const float* __restrict__ pos, float* __restrict__ out)
{
  int t  = blockIdx.x;
  int s  = t & (SLEN - 1);
  int id = ids[t];
  const float* e = emb + (size_t)id * DMODEL;
  const float* p = pos + (size_t)s * DMODEL;
  float* o = out + (size_t)t * DMODEL;
  for (int c = threadIdx.x; c < DMODEL; c += blockDim.x)
    o[c] = e[c] + p[c];
}

// ----------------------------------------------------------------------
// LayerNorm over D=512 (f32 in, f16 out for WMMA consumers)
// ----------------------------------------------------------------------
__global__ __launch_bounds__(128)
void layernorm_kernel(const float* __restrict__ in, const float* __restrict__ g,
                      const float* __restrict__ b, _Float16* __restrict__ out)
{
  __shared__ float red[128];
  int row = blockIdx.x;
  int tid = threadIdx.x;
  const float* x = in + (size_t)row * DMODEL;
  _Float16* o = out + (size_t)row * DMODEL;

  float vals[4];
  float s = 0.f;
#pragma unroll
  for (int j = 0; j < 4; ++j) { vals[j] = x[tid + j * 128]; s += vals[j]; }
  red[tid] = s; __syncthreads();
  for (int w = 64; w > 0; w >>= 1) { if (tid < w) red[tid] += red[tid + w]; __syncthreads(); }
  float mean = red[0] * (1.f / DMODEL);
  __syncthreads();

  float vs = 0.f;
#pragma unroll
  for (int j = 0; j < 4; ++j) { float d = vals[j] - mean; vs += d * d; }
  red[tid] = vs; __syncthreads();
  for (int w = 64; w > 0; w >>= 1) { if (tid < w) red[tid] += red[tid + w]; __syncthreads(); }
  float rstd = rsqrtf(red[0] * (1.f / DMODEL) + 1e-5f);

#pragma unroll
  for (int j = 0; j < 4; ++j) {
    int c = tid + j * 128;
    o[c] = (_Float16)((vals[j] - mean) * rstd * g[c] + b[c]);
  }
}

// ----------------------------------------------------------------------
// Masked softmax over LK = SLEN keys, in-place on f16 scores (f32 math).
// row r = (b*H + h)*SLEN + q. Masks: key padding (ids==0), causal.
// ----------------------------------------------------------------------
__global__ __launch_bounds__(256)
void masked_softmax_kernel(_Float16* __restrict__ scores,
                           const int* __restrict__ key_ids, int causal)
{
  __shared__ float red[256];
  int r    = blockIdx.x;
  int qpos = r & (SLEN - 1);
  int z    = r >> 10;          // /SLEN
  int b    = z >> 3;           // /NHEAD
  _Float16* row = scores + (size_t)r * SLEN;
  const int* kid = key_ids + (size_t)b * SLEN;
  int tid = threadIdx.x;

  float v[4];
  float mx = -3.4e38f;
#pragma unroll
  for (int j = 0; j < 4; ++j) {
    int kk = tid + j * 256;
    float xv = (float)row[kk];
    bool masked = (kid[kk] == 0) || (causal && (kk > qpos));
    v[j] = masked ? -3.4e38f : xv;
    mx = fmaxf(mx, v[j]);
  }
  red[tid] = mx; __syncthreads();
  for (int w = 128; w > 0; w >>= 1) { if (tid < w) red[tid] = fmaxf(red[tid], red[tid + w]); __syncthreads(); }
  mx = red[0];
  __syncthreads();

  float s = 0.f;
#pragma unroll
  for (int j = 0; j < 4; ++j) { v[j] = __expf(v[j] - mx); s += v[j]; }
  red[tid] = s; __syncthreads();
  for (int w = 128; w > 0; w >>= 1) { if (tid < w) red[tid] += red[tid + w]; __syncthreads(); }
  float inv = 1.f / red[0];
#pragma unroll
  for (int j = 0; j < 4; ++j) row[tid + j * 256] = (_Float16)(v[j] * inv);
}

// ----------------------------------------------------------------------
// Host-side helpers
// ----------------------------------------------------------------------
static inline void gemm(hipStream_t st, const _Float16* A, const _Float16* Bm,
                        const float* bias, const float* Res, float* Cf, _Float16* Ch,
                        int M, int N, int K, int lda, int ldb, int ldc,
                        int bmode, int batches, int transB, int act, float alpha)
{
  dim3 grid(N / BN, M / BM, batches);
  gemm_wmma_kernel<<<grid, 256, 0, st>>>(A, Bm, bias, Res, Cf, Ch, M, N, K,
                                         lda, ldb, ldc, bmode, transB, act, alpha);
}

extern "C" void kernel_launch(void* const* d_in, const int* in_sizes, int n_in,
                              void* d_out, int out_size, void* d_ws, size_t ws_size,
                              hipStream_t stream)
{
  (void)in_sizes; (void)n_in; (void)out_size; (void)ws_size;
#define FP(i) ((const float*)d_in[i])
  const int* src_ids = (const int*)d_in[0];
  const int* tgt_ids = (const int*)d_in[1];
  const float* src_emb = FP(2), * tgt_emb = FP(3);
  const float* src_pos = FP(4), * tgt_pos = FP(5);
  const float* e_ln1_g = FP(6),  * e_ln1_b = FP(7);
  const float* e_wq = FP(8), * e_wk = FP(9), * e_wv = FP(10), * e_wo = FP(11);
  const float* e_ln2_g = FP(12), * e_ln2_b = FP(13);
  const float* e_fc1w = FP(14), * e_fc1b = FP(15);
  const float* e_fc2w = FP(16), * e_fc2b = FP(17);
  const float* d_ln1_g = FP(18), * d_ln1_b = FP(19);
  const float* d_sawq = FP(20), * d_sawk = FP(21), * d_sawv = FP(22), * d_sawo = FP(23);
  const float* d_ln2_g = FP(24), * d_ln2_b = FP(25);
  const float* d_cawq = FP(26), * d_cawk = FP(27), * d_cawv = FP(28), * d_cawo = FP(29);
  const float* d_ln3_g = FP(30), * d_ln3_b = FP(31);
  const float* d_fc1w = FP(32), * d_fc1b = FP(33);
  const float* d_fc2w = FP(34), * d_fc2b = FP(35);
  const float* enc_lnf_g = FP(36), * enc_lnf_b = FP(37);
  const float* dec_lnf_g = FP(38), * dec_lnf_b = FP(39);
  const float* lm_head = FP(40);
#undef FP
  float* logits = (float*)d_out;

  // ---- workspace partition (256B aligned) ----
  char* base = (char*)d_ws;
  size_t off = 0;
  auto takeF = [&](size_t n) -> float* {
    float* p = (float*)(base + off); off = (off + n * 4 + 255) & ~(size_t)255; return p; };
  auto takeH = [&](size_t n) -> _Float16* {
    _Float16* p = (_Float16*)(base + off); off = (off + n * 2 + 255) & ~(size_t)255; return p; };

  const size_t NDD = (size_t)NLAYER * DMODEL * DMODEL;
  const size_t NDF = (size_t)NLAYER * DMODEL * DFF;

  float*    x    = takeF((size_t)NTOK * DMODEL);     // f32 residual stream
  _Float16* y    = takeH((size_t)NTOK * DMODEL);     // LN output (GEMM A)
  _Float16* q    = takeH((size_t)NTOK * DMODEL);
  _Float16* kbuf = takeH((size_t)NTOK * DMODEL);
  _Float16* vbuf = takeH((size_t)NTOK * DMODEL);
  _Float16* ctx  = takeH((size_t)NTOK * DMODEL);
  _Float16* ffh  = takeH((size_t)NTOK * DFF);
  _Float16* mem  = takeH((size_t)NTOK * DMODEL);     // encoder memory
  _Float16* sc   = takeH((size_t)MB * NHEAD * SLEN * SLEN);  // scores/probs (in-place)
  // f16 weights (converted once per launch)
  _Float16* h_ewq = takeH(NDD), * h_ewk = takeH(NDD), * h_ewv = takeH(NDD), * h_ewo = takeH(NDD);
  _Float16* h_ef1 = takeH(NDF), * h_ef2 = takeH(NDF);
  _Float16* h_sawq = takeH(NDD), * h_sawk = takeH(NDD), * h_sawv = takeH(NDD), * h_sawo = takeH(NDD);
  _Float16* h_cawq = takeH(NDD), * h_cawk = takeH(NDD), * h_cawv = takeH(NDD), * h_cawo = takeH(NDD);
  _Float16* h_df1 = takeH(NDF), * h_df2 = takeH(NDF);
  _Float16* h_lm  = takeH((size_t)DMODEL * NVOCAB);

  auto cvt = [&](const float* s, _Float16* d, long long n) {
    cast_f16_kernel<<<dim3((unsigned)((n + 255) / 256)), 256, 0, stream>>>(s, d, n); };
  cvt(e_wq, h_ewq, NDD);  cvt(e_wk, h_ewk, NDD);  cvt(e_wv, h_ewv, NDD);  cvt(e_wo, h_ewo, NDD);
  cvt(e_fc1w, h_ef1, NDF); cvt(e_fc2w, h_ef2, NDF);
  cvt(d_sawq, h_sawq, NDD); cvt(d_sawk, h_sawk, NDD); cvt(d_sawv, h_sawv, NDD); cvt(d_sawo, h_sawo, NDD);
  cvt(d_cawq, h_cawq, NDD); cvt(d_cawk, h_cawk, NDD); cvt(d_cawv, h_cawv, NDD); cvt(d_cawo, h_cawo, NDD);
  cvt(d_fc1w, h_df1, NDF); cvt(d_fc2w, h_df2, NDF);
  cvt(lm_head, h_lm, (long long)DMODEL * NVOCAB);

  const int ZB = MB * NHEAD;           // 32 attention batches
  const float iscale = 0.125f;         // 1/sqrt(DHEAD)
  const int D = DMODEL;

  // ================= encoder =================
  embed_kernel<<<NTOK, 256, 0, stream>>>(src_ids, src_emb, src_pos, x);
  for (int i = 0; i < NLAYER; ++i) {
    size_t wO = (size_t)i * D * D, vO = (size_t)i * D;
    size_t f1 = (size_t)i * D * DFF, b1 = (size_t)i * DFF, f2 = (size_t)i * DFF * D;

    layernorm_kernel<<<NTOK, 128, 0, stream>>>(x, e_ln1_g + vO, e_ln1_b + vO, y);
    gemm(stream, y, h_ewq + wO, nullptr, nullptr, nullptr, q,    NTOK, D, D, D, D, D, 0, 1, 0, 0, 1.f);
    gemm(stream, y, h_ewk + wO, nullptr, nullptr, nullptr, kbuf, NTOK, D, D, D, D, D, 0, 1, 0, 0, 1.f);
    gemm(stream, y, h_ewv + wO, nullptr, nullptr, nullptr, vbuf, NTOK, D, D, D, D, D, 0, 1, 0, 0, 1.f);
    gemm(stream, q, kbuf, nullptr, nullptr, nullptr, sc, SLEN, SLEN, DHEAD, D, D, SLEN, 1, ZB, 1, 0, iscale);
    masked_softmax_kernel<<<ZB * SLEN, 256, 0, stream>>>(sc, src_ids, 0);
    gemm(stream, sc, vbuf, nullptr, nullptr, nullptr, ctx, SLEN, DHEAD, SLEN, SLEN, D, D, 2, ZB, 0, 0, 1.f);
    gemm(stream, ctx, h_ewo + wO, nullptr, x, x, nullptr, NTOK, D, D, D, D, D, 0, 1, 0, 0, 1.f);

    layernorm_kernel<<<NTOK, 128, 0, stream>>>(x, e_ln2_g + vO, e_ln2_b + vO, y);
    gemm(stream, y, h_ef1 + f1, e_fc1b + b1, nullptr, nullptr, ffh, NTOK, DFF, D, D, DFF, DFF, 0, 1, 0, 1, 1.f);
    gemm(stream, ffh, h_ef2 + f2, e_fc2b + vO, x, x, nullptr, NTOK, D, DFF, DFF, D, D, 0, 1, 0, 0, 1.f);
  }
  layernorm_kernel<<<NTOK, 128, 0, stream>>>(x, enc_lnf_g, enc_lnf_b, mem);

  // ================= decoder =================
  embed_kernel<<<NTOK, 256, 0, stream>>>(tgt_ids, tgt_emb, tgt_pos, x);
  for (int i = 0; i < NLAYER; ++i) {
    size_t wO = (size_t)i * D * D, vO = (size_t)i * D;
    size_t f1 = (size_t)i * D * DFF, b1 = (size_t)i * DFF, f2 = (size_t)i * DFF * D;

    // causal self-attention
    layernorm_kernel<<<NTOK, 128, 0, stream>>>(x, d_ln1_g + vO, d_ln1_b + vO, y);
    gemm(stream, y, h_sawq + wO, nullptr, nullptr, nullptr, q,    NTOK, D, D, D, D, D, 0, 1, 0, 0, 1.f);
    gemm(stream, y, h_sawk + wO, nullptr, nullptr, nullptr, kbuf, NTOK, D, D, D, D, D, 0, 1, 0, 0, 1.f);
    gemm(stream, y, h_sawv + wO, nullptr, nullptr, nullptr, vbuf, NTOK, D, D, D, D, D, 0, 1, 0, 0, 1.f);
    gemm(stream, q, kbuf, nullptr, nullptr, nullptr, sc, SLEN, SLEN, DHEAD, D, D, SLEN, 1, ZB, 1, 0, iscale);
    masked_softmax_kernel<<<ZB * SLEN, 256, 0, stream>>>(sc, tgt_ids, 1);
    gemm(stream, sc, vbuf, nullptr, nullptr, nullptr, ctx, SLEN, DHEAD, SLEN, SLEN, D, D, 2, ZB, 0, 0, 1.f);
    gemm(stream, ctx, h_sawo + wO, nullptr, x, x, nullptr, NTOK, D, D, D, D, D, 0, 1, 0, 0, 1.f);

    // cross-attention (kv from encoder memory)
    layernorm_kernel<<<NTOK, 128, 0, stream>>>(x, d_ln2_g + vO, d_ln2_b + vO, y);
    gemm(stream, y,   h_cawq + wO, nullptr, nullptr, nullptr, q,    NTOK, D, D, D, D, D, 0, 1, 0, 0, 1.f);
    gemm(stream, mem, h_cawk + wO, nullptr, nullptr, nullptr, kbuf, NTOK, D, D, D, D, D, 0, 1, 0, 0, 1.f);
    gemm(stream, mem, h_cawv + wO, nullptr, nullptr, nullptr, vbuf, NTOK, D, D, D, D, D, 0, 1, 0, 0, 1.f);
    gemm(stream, q, kbuf, nullptr, nullptr, nullptr, sc, SLEN, SLEN, DHEAD, D, D, SLEN, 1, ZB, 1, 0, iscale);
    masked_softmax_kernel<<<ZB * SLEN, 256, 0, stream>>>(sc, src_ids, 0);
    gemm(stream, sc, vbuf, nullptr, nullptr, nullptr, ctx, SLEN, DHEAD, SLEN, SLEN, D, D, 2, ZB, 0, 0, 1.f);
    gemm(stream, ctx, h_cawo + wO, nullptr, x, x, nullptr, NTOK, D, D, D, D, D, 0, 1, 0, 0, 1.f);

    // feed-forward
    layernorm_kernel<<<NTOK, 128, 0, stream>>>(x, d_ln3_g + vO, d_ln3_b + vO, y);
    gemm(stream, y, h_df1 + f1, d_fc1b + b1, nullptr, nullptr, ffh, NTOK, DFF, D, D, DFF, DFF, 0, 1, 0, 1, 1.f);
    gemm(stream, ffh, h_df2 + f2, d_fc2b + vO, x, x, nullptr, NTOK, D, DFF, DFF, D, D, 0, 1, 0, 0, 1.f);
  }
  layernorm_kernel<<<NTOK, 128, 0, stream>>>(x, dec_lnf_g, dec_lnf_b, y);

  // ================= lm head (dominant GEMM, ~134 GFLOP) =================
  gemm(stream, y, h_lm, nullptr, nullptr, logits, nullptr,
       NTOK, NVOCAB, DMODEL, DMODEL, NVOCAB, NVOCAB, 0, 1, 0, 0, 1.f);
}
